// Rwkv6SelfAttention_40295383171544
// MI455X (gfx1250) — compile-verified, matrix operand order checked by reading, not run on hardware
//
#include <hip/hip_runtime.h>
#include <hip/hip_bf16.h>

// ---------------- problem constants (from reference) ----------------
#define BB   2
#define TT   2048
#define CC   2048
#define AA   2048
#define HH   32
#define HSZ  64
#define DMIX 32
#define DDEC 64
#define BT   (BB * TT)          // 4096 token rows
#define NMIX5 (DMIX * 5)        // 160
#define GN_EPS 6.4e-4f          // 1e-5 * 8^2

typedef __attribute__((ext_vector_type(16))) __bf16 v16bf;
typedef __attribute__((ext_vector_type(8)))  float  v8f;

// ---------------- helpers ----------------
__device__ __forceinline__ unsigned short f2bf(float f) {
  unsigned int u = __float_as_uint(f);
  u += 0x7fffu + ((u >> 16) & 1u);   // round-to-nearest-even
  return (unsigned short)(u >> 16);
}
__device__ __forceinline__ float bf2f(unsigned short s) {
  return __uint_as_float(((unsigned int)s) << 16);
}

// CDNA5 async copy: global -> LDS direct, tracked by ASYNCcnt (no VGPR transit).
// LDS operand is the low 32 bits of the shared pointer's flat address
// (LDS aperture keeps the byte offset in addr[31:0] per the ISA).
__device__ __forceinline__ void async_ld_b128(unsigned lds_off, const void* gaddr) {
  asm volatile("global_load_async_to_lds_b128 %0, %1, off"
               :: "v"(lds_off), "v"(gaddr)
               : "memory");
}
__device__ __forceinline__ void wait_async0() {
  asm volatile("s_wait_asynccnt 0x0" ::: "memory");
}

// ---------------- fp32 -> bf16 bulk convert ----------------
__global__ void k_f32_to_bf16(const float* __restrict__ src,
                              unsigned short* __restrict__ dst, int n) {
  for (int i = blockIdx.x * blockDim.x + threadIdx.x; i < n;
       i += gridDim.x * blockDim.x)
    dst[i] = f2bf(src[i]);
}

// ---------------- token shift + z = tanh(xxx @ w1)  [BT,160] ----------------
__global__ __launch_bounds__(160) void k_token_shift_z(
    const float* __restrict__ hidden, const float* __restrict__ maa_x,
    const float* __restrict__ w1 /*[C,160]*/, float* __restrict__ z) {
  __shared__ float xrow[CC];
  int bt = blockIdx.x;
  int t  = bt % TT;
  const float* h  = hidden + (size_t)bt * CC;
  const float* hp = h - CC;
  for (int c = threadIdx.x; c < CC; c += blockDim.x) {
    float hv = h[c];
    float pv = (t > 0) ? hp[c] : 0.f;
    xrow[c]  = hv + (pv - hv) * maa_x[c];
  }
  __syncthreads();
  int d = threadIdx.x;              // blockDim == 160 (5 waves)
  float acc = 0.f;
  for (int c = 0; c < CC; ++c) acc += xrow[c] * w1[c * NMIX5 + d];
  z[(size_t)bt * NMIX5 + d] = tanhf(acc);
}

// ---------------- mix: x{w,k,v,r,g} = h + xx*(maa + z@w2), stored bf16 ------
__global__ __launch_bounds__(256) void k_mix(
    const float* __restrict__ hidden, const float* __restrict__ z,
    const float* __restrict__ w2 /*[5,32,C]*/,
    const float* __restrict__ maa_w, const float* __restrict__ maa_k,
    const float* __restrict__ maa_v, const float* __restrict__ maa_r,
    const float* __restrict__ maa_g,
    unsigned short* __restrict__ xw, unsigned short* __restrict__ xk,
    unsigned short* __restrict__ xv, unsigned short* __restrict__ xr,
    unsigned short* __restrict__ xg) {
  __shared__ float zrow[NMIX5];
  int bt = blockIdx.x;
  int t  = bt % TT;
  if (threadIdx.x < NMIX5) zrow[threadIdx.x] = z[(size_t)bt * NMIX5 + threadIdx.x];
  __syncthreads();
  const float* h  = hidden + (size_t)bt * CC;
  const float* hp = h - CC;
  for (int c = threadIdx.x; c < CC; c += blockDim.x) {
    float hv = h[c];
    float xx = ((t > 0) ? hp[c] : 0.f) - hv;
    float mixv[5];
#pragma unroll
    for (int f = 0; f < 5; ++f) {
      float m = 0.f;
#pragma unroll
      for (int d = 0; d < DMIX; ++d)
        m += zrow[f * DMIX + d] * w2[(size_t)(f * DMIX + d) * CC + c];
      mixv[f] = m;
    }
    size_t o = (size_t)bt * CC + c;
    xw[o] = f2bf(hv + xx * (maa_w[c] + mixv[0]));
    xk[o] = f2bf(hv + xx * (maa_k[c] + mixv[1]));
    xv[o] = f2bf(hv + xx * (maa_v[c] + mixv[2]));
    xr[o] = f2bf(hv + xx * (maa_r[c] + mixv[3]));
    xg[o] = f2bf(hv + xx * (maa_g[c] + mixv[4]));
  }
}

// ---------------- data-dependent decay: w = exp(-exp(lowrank(xw))) ----------
__global__ __launch_bounds__(256) void k_decay(
    const unsigned short* __restrict__ xw /*bf16 [BT,C]*/,
    const float* __restrict__ dw1 /*[C,64]*/,
    const float* __restrict__ dw2 /*[64,A]*/,
    const float* __restrict__ tdecay /*[A]*/, float* __restrict__ wdec) {
  __shared__ float xrow[CC];
  __shared__ float tmp[DDEC];
  int bt = blockIdx.x;
  const unsigned short* x = xw + (size_t)bt * CC;
  for (int c = threadIdx.x; c < CC; c += blockDim.x) xrow[c] = bf2f(x[c]);
  __syncthreads();
  if (threadIdx.x < DDEC) {
    int d = threadIdx.x;
    float acc = 0.f;
    for (int c = 0; c < CC; ++c) acc += xrow[c] * dw1[c * DDEC + d];
    tmp[d] = tanhf(acc);
  }
  __syncthreads();
  for (int a = threadIdx.x; a < AA; a += blockDim.x) {
    float acc = tdecay[a];
#pragma unroll
    for (int d = 0; d < DDEC; ++d) acc += tmp[d] * dw2[d * AA + a];
    wdec[(size_t)bt * AA + a] = expf(-expf(acc));
  }
}

// ---------------- WMMA bf16 GEMM: Y[M,N] = X[M,K] * W[N,K]^T ----------------
// Double-buffered LDS, async global->LDS staging overlapped with WMMA.
#define BM 128
#define BN 128
#define BK 32

union FragU { uint4 q[2]; v16bf v; };

__global__ __launch_bounds__(256) void k_gemm_bf16(
    const unsigned short* __restrict__ X, const unsigned short* __restrict__ W,
    float* __restrict__ Y, int M, int N, int K, int epi /*0=none,1=silu*/) {
  __shared__ unsigned short Xs[2][BM * BK];
  __shared__ unsigned short Ws[2][BN * BK];

  const int tid   = threadIdx.x;
  const int mBase = blockIdx.y * BM;
  const int nBase = blockIdx.x * BN;
  const int wid   = tid >> 5, lane = tid & 31;
  const int wm = wid >> 2, wn = wid & 3;      // 2x4 wave grid -> 64x32 per wave
  const int lr = lane & 15, hi = lane >> 4;   // lane row / K-half select

  v8f acc[4][2];
#pragma unroll
  for (int i = 0; i < 4; ++i)
#pragma unroll
    for (int j = 0; j < 2; ++j)
#pragma unroll
      for (int e = 0; e < 8; ++e) acc[i][j][e] = 0.f;

  // staging map: 256 threads x 4 async b128 -> 16KB (both tiles) per K-step
  const int lrow   = tid >> 2;        // 0..63
  const int lchunk = (tid & 3) * 8;   // ushort offset within 32-wide row

  const unsigned short* xg0 = X + (size_t)(mBase + lrow) * K + lchunk;
  const unsigned short* wg0 = W + (size_t)(nBase + lrow) * K + lchunk;

  const unsigned lx0[2] = {(unsigned)(size_t)&Xs[0][lrow * BK + lchunk],
                           (unsigned)(size_t)&Xs[1][lrow * BK + lchunk]};
  const unsigned lx1[2] = {(unsigned)(size_t)&Xs[0][(lrow + 64) * BK + lchunk],
                           (unsigned)(size_t)&Xs[1][(lrow + 64) * BK + lchunk]};
  const unsigned lw0[2] = {(unsigned)(size_t)&Ws[0][lrow * BK + lchunk],
                           (unsigned)(size_t)&Ws[1][lrow * BK + lchunk]};
  const unsigned lw1[2] = {(unsigned)(size_t)&Ws[0][(lrow + 64) * BK + lchunk],
                           (unsigned)(size_t)&Ws[1][(lrow + 64) * BK + lchunk]};

  auto stage = [&](int buf, int kb) {
    async_ld_b128(lx0[buf], xg0 + kb);
    async_ld_b128(lx1[buf], xg0 + kb + (size_t)64 * K);
    async_ld_b128(lw0[buf], wg0 + kb);
    async_ld_b128(lw1[buf], wg0 + kb + (size_t)64 * K);
  };

  stage(0, 0);
  wait_async0();
  __syncthreads();

  int buf = 0;
  for (int kb = 0; kb < K; kb += BK) {
    const bool more = (kb + BK) < K;
    if (more) stage(buf ^ 1, kb + BK);   // async: overlaps with WMMAs below

    const unsigned short* xs = &Xs[buf][0];
    const unsigned short* wsld = &Ws[buf][0];
    FragU a[4], b[2];
#pragma unroll
    for (int i = 0; i < 4; ++i) {
      int r = wm * 64 + i * 16 + lr;
      a[i].q[0] = *(const uint4*)&xs[r * BK + hi * 8];
      a[i].q[1] = *(const uint4*)&xs[r * BK + 16 + hi * 8];
    }
#pragma unroll
    for (int j = 0; j < 2; ++j) {
      int r = wn * 32 + j * 16 + lr;
      b[j].q[0] = *(const uint4*)&wsld[r * BK + hi * 8];
      b[j].q[1] = *(const uint4*)&wsld[r * BK + 16 + hi * 8];
    }
#pragma unroll
    for (int i = 0; i < 4; ++i)
#pragma unroll
      for (int j = 0; j < 2; ++j)
        acc[i][j] = __builtin_amdgcn_wmma_f32_16x16x32_bf16(
            false, a[i].v, false, b[j].v, (short)0, acc[i][j], false, false);

    if (more) wait_async0();  // next tile landed in LDS
    __syncthreads();          // all waves done reading buf / next buf visible
    buf ^= 1;
  }

#pragma unroll
  for (int i = 0; i < 4; ++i)
#pragma unroll
    for (int j = 0; j < 2; ++j) {
      int gm0 = mBase + wm * 64 + i * 16 + hi * 8;   // C/D layout: M = vgpr + 8*hi
      int gn0 = nBase + wn * 32 + j * 16 + lr;       //             N = lane&15
#pragma unroll
      for (int e = 0; e < 8; ++e) {
        float val = acc[i][j][e];
        if (epi == 1) val = val / (1.f + expf(-val));  // fused SiLU (for g)
        Y[(size_t)(gm0 + e) * N + gn0] = val;
      }
    }
}

// ---------------- sequential WKV scan: one block per (b,h) ------------------
__global__ __launch_bounds__(256) void k_scan(
    const float* __restrict__ r, const float* __restrict__ k,
    const float* __restrict__ v, const float* __restrict__ wdec,
    const float* __restrict__ u /*[H,HS]*/, float* __restrict__ y /*[BT,A]*/) {
  __shared__ float rs[HSZ], ks[HSZ], ds[HSZ], vs[HSZ], us[HSZ];
  __shared__ float red[4][HSZ];
  int bh = blockIdx.x;
  int b = bh / HH, h = bh % HH;
  int tid = threadIdx.x;
  int m = tid & 63, ng = tid >> 6;  // 4 n-groups x 64 m-lanes
  if (tid < HSZ) us[tid] = u[h * HSZ + tid];
  float st[16];
#pragma unroll
  for (int i = 0; i < 16; ++i) st[i] = 0.f;
  const int nb = ng * 16;
  __syncthreads();
  for (int t = 0; t < TT; ++t) {
    size_t base = ((size_t)(b * TT + t)) * AA + h * HSZ;
    if (ng == 0)      rs[m] = r[base + m];
    else if (ng == 1) ks[m] = k[base + m];
    else if (ng == 2) ds[m] = wdec[base + m];
    else              vs[m] = v[base + m];
    __syncthreads();
    float vm  = vs[m];
    float acc = 0.f;
#pragma unroll
    for (int i = 0; i < 16; ++i) {
      int n = nb + i;
      float kk = ks[n];
      acc  += rs[n] * (us[n] * kk * vm + st[i]);
      st[i] = kk * vm + ds[n] * st[i];
    }
    red[ng][m] = acc;
    __syncthreads();
    if (ng == 0)
      y[base + m] = red[0][m] + red[1][m] + red[2][m] + red[3][m];
  }
}

// ---------------- per-head GroupNorm * gamma + beta, * silu(g), -> bf16 -----
__global__ __launch_bounds__(256) void k_gn(
    const float* __restrict__ y, const float* __restrict__ g,
    const float* __restrict__ gamma, const float* __restrict__ beta,
    unsigned short* __restrict__ ybf) {
  __shared__ float s[256];
  __shared__ float mv[4][2];
  int bt = blockIdx.x;
  int hh = threadIdx.x >> 6, m = threadIdx.x & 63;
  for (int gblk = 0; gblk < HH / 4; ++gblk) {
    int h = gblk * 4 + hh;
    int c = h * HSZ + m;
    float val = y[(size_t)bt * AA + c];
    s[threadIdx.x] = val;
    __syncthreads();
    if (m == 0) {
      float sum = 0.f, sq = 0.f;
      for (int i = 0; i < HSZ; ++i) { float x = s[hh * 64 + i]; sum += x; sq += x * x; }
      float mean = sum * (1.f / HSZ);
      float var  = sq * (1.f / HSZ) - mean * mean;
      mv[hh][0] = mean;
      mv[hh][1] = rsqrtf(var + GN_EPS);
    }
    __syncthreads();
    float o = (val - mv[hh][0]) * mv[hh][1];
    o = o * gamma[c] + beta[c];
    o *= g[(size_t)bt * AA + c];
    ybf[(size_t)bt * AA + c] = f2bf(o);
    __syncthreads();
  }
}

// ---------------- driver ----------------
extern "C" void kernel_launch(void* const* d_in, const int* in_sizes, int n_in,
                              void* d_out, int out_size, void* d_ws, size_t ws_size,
                              hipStream_t stream) {
  (void)in_sizes; (void)n_in; (void)out_size; (void)ws_size;
  const float* hidden = (const float*)d_in[0];
  const float* maa_x  = (const float*)d_in[1];
  const float* maa_w  = (const float*)d_in[2];
  const float* maa_k  = (const float*)d_in[3];
  const float* maa_v  = (const float*)d_in[4];
  const float* maa_r  = (const float*)d_in[5];
  const float* maa_g  = (const float*)d_in[6];
  const float* maa_w1 = (const float*)d_in[7];
  const float* maa_w2 = (const float*)d_in[8];
  const float* tdecay = (const float*)d_in[9];
  const float* dw1    = (const float*)d_in[10];
  const float* dw2    = (const float*)d_in[11];
  const float* faaaa  = (const float*)d_in[12];
  const float* Wr     = (const float*)d_in[13];
  const float* Wk     = (const float*)d_in[14];
  const float* Wv     = (const float*)d_in[15];
  const float* Wg     = (const float*)d_in[16];
  const float* Wo     = (const float*)d_in[17];
  const float* gamma  = (const float*)d_in[18];
  const float* beta   = (const float*)d_in[19];

  char* ws = (char*)d_ws;
  size_t off = 0;
  auto take = [&](size_t bytes) -> char* {
    char* p = ws + off;
    off = (off + bytes + 255) & ~(size_t)255;
    return p;
  };

  unsigned short* wr_bf = (unsigned short*)take((size_t)AA * CC * 2);
  unsigned short* wk_bf = (unsigned short*)take((size_t)AA * CC * 2);
  unsigned short* wv_bf = (unsigned short*)take((size_t)AA * CC * 2);
  unsigned short* wg_bf = (unsigned short*)take((size_t)AA * CC * 2);
  unsigned short* wo_bf = (unsigned short*)take((size_t)CC * AA * 2);
  unsigned short* xw_bf = (unsigned short*)take((size_t)BT * CC * 2);
  unsigned short* xk_bf = (unsigned short*)take((size_t)BT * CC * 2);
  unsigned short* xv_bf = (unsigned short*)take((size_t)BT * CC * 2);
  unsigned short* xr_bf = (unsigned short*)take((size_t)BT * CC * 2);
  unsigned short* xg_bf = (unsigned short*)take((size_t)BT * CC * 2);
  float* zbuf  = (float*)take((size_t)BT * NMIX5 * 4);
  float* rbuf  = (float*)take((size_t)BT * AA * 4);
  float* kbuf  = (float*)take((size_t)BT * AA * 4);
  float* vbuf  = (float*)take((size_t)BT * AA * 4);
  float* gbuf  = (float*)take((size_t)BT * AA * 4);
  float* wdecb = (float*)take((size_t)BT * AA * 4);
  float* ybuf  = (float*)take((size_t)BT * AA * 4);
  unsigned short* ybf = (unsigned short*)take((size_t)BT * AA * 2);

  const int nW = AA * CC;
  k_f32_to_bf16<<<2048, 256, 0, stream>>>(Wr, wr_bf, nW);
  k_f32_to_bf16<<<2048, 256, 0, stream>>>(Wk, wk_bf, nW);
  k_f32_to_bf16<<<2048, 256, 0, stream>>>(Wv, wv_bf, nW);
  k_f32_to_bf16<<<2048, 256, 0, stream>>>(Wg, wg_bf, nW);
  k_f32_to_bf16<<<2048, 256, 0, stream>>>(Wo, wo_bf, nW);

  k_token_shift_z<<<BT, 160, 0, stream>>>(hidden, maa_x, maa_w1, zbuf);
  k_mix<<<BT, 256, 0, stream>>>(hidden, zbuf, maa_w2, maa_w, maa_k, maa_v,
                                maa_r, maa_g, xw_bf, xk_bf, xv_bf, xr_bf, xg_bf);
  k_decay<<<BT, 256, 0, stream>>>(xw_bf, dw1, dw2, tdecay, wdecb);

  dim3 gProj(AA / BN, BT / BM);
  k_gemm_bf16<<<gProj, 256, 0, stream>>>(xr_bf, wr_bf, rbuf, BT, AA, CC, 0);
  k_gemm_bf16<<<gProj, 256, 0, stream>>>(xk_bf, wk_bf, kbuf, BT, AA, CC, 0);
  k_gemm_bf16<<<gProj, 256, 0, stream>>>(xv_bf, wv_bf, vbuf, BT, AA, CC, 0);
  k_gemm_bf16<<<gProj, 256, 0, stream>>>(xg_bf, wg_bf, gbuf, BT, AA, CC, 1);

  k_scan<<<BB * HH, 256, 0, stream>>>(rbuf, kbuf, vbuf, wdecb, faaaa, ybuf);
  k_gn<<<BT, 256, 0, stream>>>(ybuf, gbuf, gamma, beta, ybf);

  dim3 gOut(CC / BN, BT / BM);
  k_gemm_bf16<<<gOut, 256, 0, stream>>>(ybf, wo_bf, (float*)d_out, BT, CC, AA, 0);
}
